// GCNSpatialBlock_18476949307582
// MI455X (gfx1250) — matrix operand order repaired and checked by reading, not compiled, not run on hardware
//
#include <hip/hip_runtime.h>

typedef __attribute__((ext_vector_type(16))) _Float16 v16h;
typedef __attribute__((ext_vector_type(8)))  float    v8f;

#define J     17
#define C     256
#define NTOT  15552          // 64*243
#define NPER  12
#define NBLK  1296           // NTOT / NPER
#define XST   260            // padded LDS row stride (floats); 260*4 % 16 == 0
#define MTOT  (NTOT * J)     // 264384 rows for BN stats

// ---------------------------------------------------------------------------
// Kernel 1: transpose-convert W -> Wt (f16, [c_out][k]) and build the 17x17
// static adjacency matrix S on-device (one tiny block).
// ---------------------------------------------------------------------------
__global__ void prep_kernel(const float* __restrict__ W,
                            const float* __restrict__ L1,
                            const float* __restrict__ L2,
                            const float* __restrict__ w1,
                            const float* __restrict__ w2,
                            _Float16* __restrict__ Wt,
                            float* __restrict__ S) {
  int g = blockIdx.x * 256 + threadIdx.x;
  for (int e = g; e < C * C; e += 64 * 256) {
    int c = e >> 8, k = e & 255;
    Wt[e] = (_Float16)W[k * C + c];
  }
  if (blockIdx.x == 0) {
    // CONNECTIONS as row bitmasks (directed, per reference dict)
    const unsigned int adj[J] = {0x00082u, 0x00005u, 0x0000Au, 0x00004u, 0x00021u,
                                 0x00050u, 0x00020u, 0x00101u, 0x04A80u, 0x00500u,
                                 0x00200u, 0x01100u, 0x02800u, 0x01000u, 0x08100u,
                                 0x14000u, 0x08000u};
    // stable softplus
    float sp1 = fmaxf(w1[0], 0.f) + log1pf(expf(-fabsf(w1[0])));
    float sp2 = fmaxf(w2[0], 0.f) + log1pf(expf(-fabsf(w2[0])));
    for (int p = threadIdx.x; p < J * J; p += 256) {
      int i = p / J, j = p % J;
      float f[2];
      int ii = i, jj = j;
      for (int s = 0; s < 2; ++s) {
        unsigned p2m = 0;
        for (int k = 0; k < J; ++k)
          if ((adj[ii] >> k) & 1u) p2m |= adj[k];
        float a1 = (((adj[ii] >> jj) & 1u) || (ii == jj)) ? 1.f : 0.f;
        float a2 = (((p2m >> jj) & 1u) && !((adj[ii] >> jj) & 1u) && (ii != jj)) ? 1.f : 0.f;
        float s1 = 1.f / (1.f + expf(-L1[ii * J + jj]));
        float s2 = 1.f / (1.f + expf(-L2[ii * J + jj]));
        f[s] = sp1 * (a1 + s1) + sp2 * (a2 + s2);
        int tmp = ii; ii = jj; jj = tmp;
      }
      S[p] = 0.5f * (f[0] + f[1]);   // 0.5*(S + S^T)
    }
  }
}

// ---------------------------------------------------------------------------
// Kernel 2/4: main fused pass. PASS==0: accumulate per-channel BN partials.
// PASS==1: recompute h, apply BN scale/shift, relu, residual, write out.
// One block = 256 threads = 8 waves; W (f16) resident in LDS (128 KB).
// The cosine-similarity gram AND the gate dot both ride the tensor units:
// xs row 17 holds gate_w, so G = [x; gate_w] @ x^T gives norms (diag),
// gate dots (col 17) and raw cosine numerators in 4 extra WMMA tiles.
// ---------------------------------------------------------------------------
template <int PASS>
__global__ __launch_bounds__(256, 1)
void pass_kernel(const float* __restrict__ x,
                 const _Float16* __restrict__ Wt,
                 const float* __restrict__ Sg,
                 const float* __restrict__ bias,
                 const float* __restrict__ gw,
                 const float* __restrict__ gb,
                 const float* __restrict__ stats,
                 float* __restrict__ partials,
                 float* __restrict__ out) {
  __shared__ _Float16 Wl[C * C];       // 128 KB, B operand [c_out][k]
  __shared__ float xs[18 * XST];       // rows 0..16: x tile; row 17: gate_w
  __shared__ float ys[J * XST];        // y = x@W tile (f32)
  __shared__ float sS[J * 18];
  __shared__ float sA[J * 18];         // A, then A_hat in place
  __shared__ float sG[J * 20];         // raw gram (cols 0..16) + gate dot (col 17)
  __shared__ float sInv[32];
  __shared__ float sGate[32];
  __shared__ float sDinv[32];

  const int t    = threadIdx.x;
  const int lane = t & 31;
  const int wave = t >> 5;
  const int hsel = lane >> 4;          // 0: lanes 0-15, 1: lanes 16-31
  const int lidx = lane & 15;

  // convert two 8-float chunks p0[0..7], p1[0..7] to a v16h fragment
  auto cvt16 = [](const float* p0, const float* p1) -> v16h {
    float4 c0 = *(const float4*)(p0);
    float4 c1 = *(const float4*)(p0 + 4);
    float4 c2 = *(const float4*)(p1);
    float4 c3 = *(const float4*)(p1 + 4);
    v16h a;
    a[0]=(_Float16)c0.x;  a[1]=(_Float16)c0.y;  a[2]=(_Float16)c0.z;  a[3]=(_Float16)c0.w;
    a[4]=(_Float16)c1.x;  a[5]=(_Float16)c1.y;  a[6]=(_Float16)c1.z;  a[7]=(_Float16)c1.w;
    a[8]=(_Float16)c2.x;  a[9]=(_Float16)c2.y;  a[10]=(_Float16)c2.z; a[11]=(_Float16)c2.w;
    a[12]=(_Float16)c3.x; a[13]=(_Float16)c3.y; a[14]=(_Float16)c3.z; a[15]=(_Float16)c3.w;
    return a;
  };

  // Block-resident data: W into LDS (128-bit copies), S, gate_w (xs row 17)
  {
    const uint4* wg = (const uint4*)Wt;
    uint4* wl = (uint4*)Wl;
    for (int e = t; e < (C * C) / 8; e += 256) wl[e] = wg[e];
    for (int p = t; p < J * J; p += 256) sS[(p / J) * 18 + (p % J)] = Sg[p];
    xs[17 * XST + t] = gw[t];
  }
  const float bias_c = bias[t];
  const float gateB  = gb[0];
  float scl = 0.f, shf = 0.f;
  if (PASS == 1) { scl = stats[t]; shf = stats[C + t]; }
  float cs = 0.f, cq = 0.f;            // channel t partial sums (PASS 0)
  __syncthreads();

  const int n0 = blockIdx.x * NPER;
  for (int nn = 0; nn < NPER; ++nn) {
    const int n = n0 + nn;
    const float* xg = x + (size_t)n * (J * C);
    #pragma unroll
    for (int k = 0; k < J; ++k) xs[k * XST + t] = xg[k * C + t];
    if (nn + 1 < NPER) __builtin_prefetch(xg + J * C + t * J, 0, 0);
    __syncthreads();

    // ---- main GEMM y = x @ W via v_wmma_f32_16x16x32_f16 ----
    // wave w owns N-tiles {2w, 2w+1}; M-tile0 = rows 0..15, M-tile1 = row 16
    #pragma unroll
    for (int nt2 = 0; nt2 < 2; ++nt2) {
      const int nt = wave * 2 + nt2;
      v8f acc0 = {};
      v8f acc1 = {};
      v16h a1;
      #pragma unroll
      for (int e = 0; e < 16; ++e) a1[e] = (_Float16)0.f;   // zero once, not per K
      #pragma unroll
      for (int k0 = 0; k0 < 8; ++k0) {
        const int kb = k0 * 32;
        // B fragment: lane -> col = nt*16+lidx; K chunks kb+hsel*8 and kb+16+hsel*8
        union { v16h v; uint4 u[2]; } B;
        const _Float16* wp = &Wl[(nt * 16 + lidx) * C + kb + hsel * 8];
        B.u[0] = *(const uint4*)wp;
        B.u[1] = *(const uint4*)(wp + 16);
        // A fragment, M-tile 0: row = lidx; K chunks kb+hsel*8 and kb+16+hsel*8
        const float* xr0 = &xs[lidx * XST + kb + hsel * 8];
        v16h a0 = cvt16(xr0, xr0 + 16);
        // A fragment, M-tile 1: only global row 16 (lidx==0); rest stays zero
        if (lidx == 0) {
          const float* xr1 = &xs[16 * XST + kb + hsel * 8];
          a1 = cvt16(xr1, xr1 + 16);
        }
        acc0 = __builtin_amdgcn_wmma_f32_16x16x32_f16(false, a0, false, B.v, (short)0, acc0, false, false);
        acc1 = __builtin_amdgcn_wmma_f32_16x16x32_f16(false, a1, false, B.v, (short)0, acc1, false, false);
      }
      // C/D layout: VGPR r -> row r + 8*hsel, col = nt*16 + lidx
      const int col = nt * 16 + lidx;
      #pragma unroll
      for (int r = 0; r < 8; ++r) ys[(r + 8 * hsel) * XST + col] = acc0[r];
      if (hsel == 0) ys[16 * XST + col] = acc1[0];
    }

    // ---- raw gram G = [x; gate_w] @ x^T on tensor units (waves 0..3) ----
    if (wave < 4) {
      const int gmt = wave >> 1, gnt = wave & 1;
      v8f gacc = {};
      #pragma unroll
      for (int k0 = 0; k0 < 8; ++k0) {
        const int kb = k0 * 32;
        v16h ga;
        #pragma unroll
        for (int e = 0; e < 16; ++e) ga[e] = (_Float16)0.f;
        const int ar = gmt * 16 + lidx;          // A rows: 0..15 / 16..31
        if (ar < 18) {
          const float* xr = &xs[ar * XST + kb + hsel * 8];
          ga = cvt16(xr, xr + 16);
        }
        v16h gbf;
        #pragma unroll
        for (int e = 0; e < 16; ++e) gbf[e] = (_Float16)0.f;
        const int bc = gnt * 16 + lidx;          // B cols = x rows (X^T)
        if (bc < 18) {
          const float* xr = &xs[bc * XST + kb + hsel * 8];
          gbf = cvt16(xr, xr + 16);
        }
        gacc = __builtin_amdgcn_wmma_f32_16x16x32_f16(false, ga, false, gbf, (short)0, gacc, false, false);
      }
      const int col = gnt * 16 + lidx;
      if (col < 18) {
        #pragma unroll
        for (int r = 0; r < 8; ++r) {
          const int row = gmt * 16 + r + 8 * hsel;
          if (row < J) sG[row * 20 + col] = gacc[r];
        }
      }
    }
    __syncthreads();

    // ---- per-row inv-norm (gram diag) and gate (gram col 17) ----
    if (t < J) {
      float g2 = fmaxf(sG[t * 20 + t], 0.f);
      sInv[t]  = 1.f / fmaxf(sqrtf(g2), 1e-12f);
      sGate[t] = 1.f / (1.f + expf(-(sG[t * 20 + 17] + gateB)));
    }
    __syncthreads();

    // ---- adjacency blend: A = g*S + (1-g)*(relu(cos) + I) ----
    for (int p = t; p < J * J; p += 256) {
      int i = p / J, j = p % J;
      float cosv = sG[i * 20 + j] * sInv[i] * sInv[j];
      float dyn  = fmaxf(cosv, 0.f) + ((i == j) ? 1.f : 0.f);
      float g    = sGate[i];
      sA[i * 18 + j] = g * sS[i * 18 + j] + (1.f - g) * dyn;
    }
    __syncthreads();
    if (t < J) {
      float s = 0.f;
      for (int j = 0; j < J; ++j) s += sA[t * 18 + j];
      sDinv[t] = rsqrtf(s + 1e-6f);
    }
    __syncthreads();
    for (int p = t; p < J * J; p += 256) {
      int i = p / J, j = p % J;
      sA[i * 18 + j] *= sDinv[i] * sDinv[j];   // A_hat in place
    }
    __syncthreads();

    // ---- epilogue: h = A_hat @ y + bias; thread t owns channel t ----
    float yv[J];
    #pragma unroll
    for (int j = 0; j < J; ++j) yv[j] = ys[j * XST + t];
    #pragma unroll
    for (int i = 0; i < J; ++i) {
      float h = bias_c;
      #pragma unroll
      for (int j = 0; j < J; ++j) h = fmaf(sA[i * 18 + j], yv[j], h);
      if (PASS == 0) {
        cs += h;
        cq = fmaf(h, h, cq);
      } else {
        float o = fmaxf(fmaf(h, scl, shf), 0.f) + xs[i * XST + t];
        out[(size_t)n * (J * C) + i * C + t] = o;
      }
    }
    __syncthreads();
  }

  if (PASS == 0) {
    partials[(size_t)blockIdx.x * 512 + t]       = cs;
    partials[(size_t)blockIdx.x * 512 + 256 + t] = cq;
  }
}

// ---------------------------------------------------------------------------
// Kernel 3: fixed-order reduction of BN partials -> per-channel scale/shift
// ---------------------------------------------------------------------------
__global__ void reduce_kernel(const float* __restrict__ partials,
                              const float* __restrict__ gamma,
                              const float* __restrict__ beta,
                              float* __restrict__ stats) {
  int t = threadIdx.x;
  float s = 0.f, q = 0.f;
  for (int b = 0; b < NBLK; ++b) {
    s += partials[(size_t)b * 512 + t];
    q += partials[(size_t)b * 512 + 256 + t];
  }
  const float invM = 1.f / (float)MTOT;
  float mean = s * invM;
  float var  = fmaf(q, invM, -mean * mean);     // biased var, matches jnp.var
  float rstd = rsqrtf(var + 1e-5f);
  float scl  = gamma[t] * rstd;
  stats[t]       = scl;
  stats[256 + t] = beta[t] - mean * scl;
}

// ---------------------------------------------------------------------------
extern "C" void kernel_launch(void* const* d_in, const int* in_sizes, int n_in,
                              void* d_out, int out_size, void* d_ws, size_t ws_size,
                              hipStream_t stream) {
  (void)in_sizes; (void)n_in; (void)out_size; (void)ws_size;
  const float* x     = (const float*)d_in[0];
  const float* L1    = (const float*)d_in[1];
  const float* L2    = (const float*)d_in[2];
  const float* w1    = (const float*)d_in[3];
  const float* w2    = (const float*)d_in[4];
  const float* W     = (const float*)d_in[5];
  const float* bias  = (const float*)d_in[6];
  const float* gw    = (const float*)d_in[7];
  const float* gb    = (const float*)d_in[8];
  const float* gamma = (const float*)d_in[9];
  const float* beta  = (const float*)d_in[10];
  float* out = (float*)d_out;

  // Workspace layout (float offsets): Wt_h occupies first 32768 floats (128 KB)
  _Float16* Wt   = (_Float16*)d_ws;
  float* wsf     = (float*)d_ws;
  float* Sg      = wsf + 32768;           // 289 floats
  float* stats   = wsf + 33152;           // 512 floats (scale | shift)
  float* partial = wsf + 33664;           // NBLK * 512 floats (~2.65 MB)

  prep_kernel<<<64, 256, 0, stream>>>(W, L1, L2, w1, w2, Wt, Sg);
  pass_kernel<0><<<NBLK, 256, 0, stream>>>(x, Wt, Sg, bias, gw, gb, nullptr, partial, nullptr);
  reduce_kernel<<<1, 256, 0, stream>>>(partial, gamma, beta, stats);
  pass_kernel<1><<<NBLK, 256, 0, stream>>>(x, Wt, Sg, bias, gw, gb, stats, nullptr, out);
}
